// Critic_44367012168421
// MI455X (gfx1250) — matrix operand order repaired and checked by reading, not compiled, not run on hardware
//
#include <hip/hip_runtime.h>
#include <stdint.h>

#define S_STEPS 32768
#define D_DIM   1024
#define G_DIM   4096   // 4*D

// ---------------------------------------------------------------------------
// WMMA types (CDNA5 / gfx1250, wave32)
// ---------------------------------------------------------------------------
typedef __attribute__((ext_vector_type(16))) __bf16 v16bf;
typedef __attribute__((ext_vector_type(8)))  float  v8f;

union FragBF { uint32_t u[8]; v16bf v; };

__device__ __forceinline__ unsigned int f32_to_bf16_bits(float f) {
  unsigned int u = __float_as_uint(f);
  return (u + 0x7FFFu + ((u >> 16) & 1u)) >> 16;   // round-to-nearest-even
}

// ---------------------------------------------------------------------------
// f32 -> bf16 conversion, 4 elements / thread
// ---------------------------------------------------------------------------
__global__ void cvt_f32_to_bf16_v4(const float* __restrict__ in,
                                   unsigned short* __restrict__ out, int n4) {
  int i = blockIdx.x * blockDim.x + threadIdx.x;
  if (i >= n4) return;
  float4 v = reinterpret_cast<const float4*>(in)[i];
  uint2 o;
  o.x = f32_to_bf16_bits(v.x) | (f32_to_bf16_bits(v.y) << 16);
  o.y = f32_to_bf16_bits(v.z) | (f32_to_bf16_bits(v.w) << 16);
  reinterpret_cast<uint2*>(out)[i] = o;
}

// ---------------------------------------------------------------------------
// WMMA GEMM:  C[m,n] = sum_k A[m,k] * W[n,k]  (+ bias[n])
//   A: M x K bf16 row-major,  W: N x K bf16 row-major (acts as B^T)
//   mode 0: relu + store bf16,  mode 1: store f32
// Block: 256 threads = 8 waves. Each wave: 32(M) x 64(N) output.
// ---------------------------------------------------------------------------
__global__ __launch_bounds__(256) void wmma_gemm_bf16(
    const unsigned short* __restrict__ A,
    const unsigned short* __restrict__ B,
    const float* __restrict__ bias,
    unsigned short* __restrict__ outBF,
    float* __restrict__ outF,
    int M, int N, int K, int mode) {
  const int lane = threadIdx.x & 31;
  const int wave = threadIdx.x >> 5;
  const int rt = lane & 15;    // row within 16x16 tile (A: M, B: N)
  const int kh = lane >> 4;    // lane half

  const int m0 = blockIdx.y * 256 + wave * 32;
  const int n0 = blockIdx.x * 64;

  v8f acc[2][4];
#pragma unroll
  for (int mi = 0; mi < 2; ++mi)
#pragma unroll
    for (int ni = 0; ni < 4; ++ni)
#pragma unroll
      for (int r = 0; r < 8; ++r) acc[mi][ni][r] = 0.0f;

  for (int kb = 0; kb < K; kb += 32) {
    FragBF a[2], b[4];
    // A fragment layout (16-bit, 16x32):
    //   elems 0..7  -> K = kb + kh*8 + (0..7)
    //   elems 8..15 -> K = kb + 16 + kh*8 + (0..7)
#pragma unroll
    for (int mi = 0; mi < 2; ++mi) {
      const unsigned short* pa = A + (size_t)(m0 + mi * 16 + rt) * K + kb + kh * 8;
      uint4 lo = *reinterpret_cast<const uint4*>(pa);
      uint4 hi = *reinterpret_cast<const uint4*>(pa + 16);
      a[mi].u[0] = lo.x; a[mi].u[1] = lo.y; a[mi].u[2] = lo.z; a[mi].u[3] = lo.w;
      a[mi].u[4] = hi.x; a[mi].u[5] = hi.y; a[mi].u[6] = hi.z; a[mi].u[7] = hi.w;
    }
    // B fragment layout (16-bit, 32x16), same K order as A, lane = N:
    //   elems 0..7  -> K = kb + kh*8 + (0..7), elems 8..15 -> K = kb + 16 + kh*8 + (0..7)
#pragma unroll
    for (int ni = 0; ni < 4; ++ni) {
      const unsigned short* pb = B + (size_t)(n0 + ni * 16 + rt) * K + kb + kh * 8;
      uint4 lo = *reinterpret_cast<const uint4*>(pb);
      uint4 hi = *reinterpret_cast<const uint4*>(pb + 16);
      b[ni].u[0] = lo.x; b[ni].u[1] = lo.y; b[ni].u[2] = lo.z; b[ni].u[3] = lo.w;
      b[ni].u[4] = hi.x; b[ni].u[5] = hi.y; b[ni].u[6] = hi.z; b[ni].u[7] = hi.w;
    }
#pragma unroll
    for (int mi = 0; mi < 2; ++mi)
#pragma unroll
      for (int ni = 0; ni < 4; ++ni)
        acc[mi][ni] = __builtin_amdgcn_wmma_f32_16x16x32_bf16(
            false, a[mi].v, false, b[ni].v, (short)0, acc[mi][ni], false, false);
  }

  // Epilogue. C/D layout: vgpr r -> row = r + 8*kh within tile, col = lane&15.
#pragma unroll
  for (int mi = 0; mi < 2; ++mi) {
#pragma unroll
    for (int ni = 0; ni < 4; ++ni) {
      const int col = n0 + ni * 16 + rt;
      const float bv = bias[col];
#pragma unroll
      for (int r = 0; r < 8; ++r) {
        const int row = m0 + mi * 16 + kh * 8 + r;
        float v = acc[mi][ni][r] + bv;
        if (mode == 0) {
          v = fmaxf(v, 0.0f);
          outBF[(size_t)row * N + col] = (unsigned short)f32_to_bf16_bits(v);
        } else {
          outF[(size_t)row * N + col] = v;
        }
      }
    }
  }
}

// ---------------------------------------------------------------------------
// Persistent LSTM scan. 64 workgroups x 256 threads.
// Block w owns hidden units j in [w*16, w*16+16), i.e. gate rows
// {g*1024 + j} for g in {i,f,g,o}. Its 64 x 1024 slice of W_hh lives in LDS
// as bf16 pairs (128 KB) for the whole scan -> 128 KB of LDS reads per step
// instead of 16 MB from memory.
// Cross-WGP step barrier: per-WG flag array (release store + acquire polls),
// avoiding serialized same-address atomics.
// ---------------------------------------------------------------------------
#define WROWU 520  // padded LDS row stride in uints (520 % 64 == 8 -> no bank clash)

__global__ __launch_bounds__(256) void lstm_scan(
    const unsigned int* __restrict__ WhhB,  // 4096 x 512 (bf16 pairs)
    const float* __restrict__ GX,           // S x 4096 (b_ih already added)
    const float* __restrict__ bhh,          // 4096
    float* __restrict__ HS,                 // S x 1024 output history
    unsigned int* __restrict__ flags,       // gridDim.x step flags
    int S) {
  extern __shared__ unsigned int smemu[];
  unsigned int* Wsh = smemu;                      // 64 * WROWU uints
  float* h_sh    = (float*)(Wsh + 64 * WROWU);    // 1024
  float* part    = h_sh + 1024;                   // 256
  float* gbias   = part + 256;                    // 64
  float* gate_sh = gbias + 64;                    // 64
  float* c_sh    = gate_sh + 64;                  // 16

  const int tid = threadIdx.x;
  const int wg = blockIdx.x;
  const int j0 = wg * 16;

  // Stage bf16 W_hh slice + biases into LDS (once).
  for (int idx = tid; idx < 64 * 512; idx += 256) {
    const int r = idx >> 9;          // 0..63
    const int ku = idx & 511;        // uint (= bf16 pair) index along K
    const int gate = r >> 4;
    const int jj = r & 15;
    Wsh[r * WROWU + ku] = WhhB[(size_t)(gate * D_DIM + j0 + jj) * 512 + ku];
  }
  if (tid < 64) {
    const int gate = tid >> 4, jj = tid & 15;
    gbias[tid] = bhh[gate * D_DIM + j0 + jj];
  }
  if (tid < 16) c_sh[tid] = 0.0f;
  __syncthreads();

  const int r = tid >> 2;            // 0..63: owned gate row
  const int q = tid & 3;             // quarter of K
  const int grow = (r >> 4) * D_DIM + j0 + (r & 15);  // global gate row
  const uint4* wr = reinterpret_cast<const uint4*>(Wsh + r * WROWU + q * 128);

  for (int t = 0; t < S; ++t) {
    // Prefetch this step's gx element early (hides L2 latency under the dot).
    float gxv = 0.0f;
    if (tid < 64) gxv = GX[(size_t)t * G_DIM + grow];

    // 1. Fetch previous hidden state into LDS (one float4 per thread).
    {
      float4 hv = make_float4(0.f, 0.f, 0.f, 0.f);
      if (t > 0) hv = reinterpret_cast<const float4*>(HS + (size_t)(t - 1) * D_DIM)[tid];
      reinterpret_cast<float4*>(h_sh)[tid] = hv;
    }
    __syncthreads();

    // 2. 64 dots of length 1024 (4 partial threads per row), bf16 weights.
    const float4* hr = reinterpret_cast<const float4*>(h_sh + q * 256);
    float p = 0.0f;
#pragma unroll 4
    for (int k = 0; k < 32; ++k) {
      uint4 w = wr[k];
      float4 ha = hr[2 * k];
      float4 hb = hr[2 * k + 1];
      p = fmaf(__uint_as_float(w.x << 16),          ha.x, p);
      p = fmaf(__uint_as_float(w.x & 0xffff0000u),  ha.y, p);
      p = fmaf(__uint_as_float(w.y << 16),          ha.z, p);
      p = fmaf(__uint_as_float(w.y & 0xffff0000u),  ha.w, p);
      p = fmaf(__uint_as_float(w.z << 16),          hb.x, p);
      p = fmaf(__uint_as_float(w.z & 0xffff0000u),  hb.y, p);
      p = fmaf(__uint_as_float(w.w << 16),          hb.z, p);
      p = fmaf(__uint_as_float(w.w & 0xffff0000u),  hb.w, p);
    }
    part[tid] = p;
    __syncthreads();

    // 3. Reduce partials, add gx + b_hh.
    if (tid < 64) {
      float g = part[tid * 4] + part[tid * 4 + 1] + part[tid * 4 + 2] + part[tid * 4 + 3];
      gate_sh[tid] = g + gbias[tid] + gxv;
    }
    __syncthreads();

    // 4. Gate nonlinearities + state update for owned hidden units.
    if (tid < 16) {
      const float gi = gate_sh[tid];
      const float gf = gate_sh[16 + tid];
      const float gg = gate_sh[32 + tid];
      const float go = gate_sh[48 + tid];
      const float i = 1.0f / (1.0f + __expf(-gi));
      const float f = 1.0f / (1.0f + __expf(-gf));
      const float g = tanhf(gg);
      const float o = 1.0f / (1.0f + __expf(-go));
      const float c = f * c_sh[tid] + i * g;
      c_sh[tid] = c;
      HS[(size_t)t * D_DIM + j0 + tid] = o * tanhf(c);
      __threadfence();
    }
    __syncthreads();

    // 5. Device-wide step barrier: publish own flag, poll all flags in parallel.
    if (tid == 0) {
      __hip_atomic_store(&flags[wg], (unsigned)(t + 1), __ATOMIC_RELEASE,
                         __HIP_MEMORY_SCOPE_AGENT);
    }
    if (tid < 64) {
      while (__hip_atomic_load(&flags[tid], __ATOMIC_ACQUIRE,
                               __HIP_MEMORY_SCOPE_AGENT) < (unsigned)(t + 1)) {
        __builtin_amdgcn_s_sleep(1);
      }
    }
    __syncthreads();
  }
}

// ---------------------------------------------------------------------------
// values[t] = dot(HS[t], w_head) + b_head
// ---------------------------------------------------------------------------
__global__ __launch_bounds__(256) void head_kernel(
    const float* __restrict__ HS, const float* __restrict__ wh,
    const float* __restrict__ bh, float* __restrict__ out) {
  __shared__ float red[256];
  const int row = blockIdx.x;
  const int tid = threadIdx.x;
  const float* h = HS + (size_t)row * D_DIM;
  float s = 0.0f;
  for (int k = tid; k < D_DIM; k += 256) s = fmaf(h[k], wh[k], s);
  red[tid] = s;
  __syncthreads();
  for (int off = 128; off > 0; off >>= 1) {
    if (tid < off) red[tid] += red[tid + off];
    __syncthreads();
  }
  if (tid == 0) out[row] = red[0] + bh[0];
}

__global__ void zero_u32_n(unsigned int* p, int n) {
  int i = blockIdx.x * blockDim.x + threadIdx.x;
  if (i < n) p[i] = 0u;
}

// ---------------------------------------------------------------------------
extern "C" void kernel_launch(void* const* d_in, const int* in_sizes, int n_in,
                              void* d_out, int out_size, void* d_ws, size_t ws_size,
                              hipStream_t stream) {
  (void)in_sizes; (void)n_in; (void)out_size; (void)ws_size;
  const float* obs   = (const float*)d_in[0];
  const float* Wmlp  = (const float*)d_in[1];
  const float* bmlp  = (const float*)d_in[2];
  const float* Wih   = (const float*)d_in[3];
  const float* bih   = (const float*)d_in[4];
  const float* Whh   = (const float*)d_in[5];
  const float* bhh   = (const float*)d_in[6];
  const float* whead = (const float*)d_in[7];
  const float* bhead = (const float*)d_in[8];
  float* out = (float*)d_out;

  char* ws = (char*)d_ws;
  size_t off = 0;
  auto alloc = [&](size_t bytes) -> void* {
    void* p = ws + off;
    off += (bytes + 255) & ~(size_t)255;
    return p;
  };
  unsigned int*   flags = (unsigned int*)  alloc(64 * sizeof(unsigned int));
  unsigned short* OBSb  = (unsigned short*)alloc((size_t)S_STEPS * D_DIM * 2);
  unsigned short* Wmlpb = (unsigned short*)alloc((size_t)D_DIM * D_DIM * 2);
  unsigned short* Wihb  = (unsigned short*)alloc((size_t)G_DIM * D_DIM * 2);
  unsigned short* Whhb  = (unsigned short*)alloc((size_t)G_DIM * D_DIM * 2);
  unsigned short* Xb    = (unsigned short*)alloc((size_t)S_STEPS * D_DIM * 2);
  float*          GX    = (float*)         alloc((size_t)S_STEPS * (size_t)G_DIM * 4);
  float*          HS    = (float*)         alloc((size_t)S_STEPS * D_DIM * 4);

  // bf16 conversions
  {
    int n4 = (S_STEPS * D_DIM) / 4;
    cvt_f32_to_bf16_v4<<<(n4 + 255) / 256, 256, 0, stream>>>(obs, OBSb, n4);
    n4 = (D_DIM * D_DIM) / 4;
    cvt_f32_to_bf16_v4<<<(n4 + 255) / 256, 256, 0, stream>>>(Wmlp, Wmlpb, n4);
    n4 = (G_DIM * D_DIM) / 4;
    cvt_f32_to_bf16_v4<<<(n4 + 255) / 256, 256, 0, stream>>>(Wih, Wihb, n4);
    cvt_f32_to_bf16_v4<<<(n4 + 255) / 256, 256, 0, stream>>>(Whh, Whhb, n4);
  }

  // GEMM1: X = relu(obs @ Wmlp^T + b_mlp), bf16 out
  {
    dim3 grid(D_DIM / 64, S_STEPS / 256);
    wmma_gemm_bf16<<<grid, 256, 0, stream>>>(OBSb, Wmlpb, bmlp, Xb, nullptr,
                                             S_STEPS, D_DIM, D_DIM, 0);
  }
  // GEMM2: GX = X @ Wih^T + b_ih, f32 out
  {
    dim3 grid(G_DIM / 64, S_STEPS / 256);
    wmma_gemm_bf16<<<grid, 256, 0, stream>>>(Xb, Wihb, bih, nullptr, GX,
                                             S_STEPS, G_DIM, D_DIM, 1);
  }

  // Persistent LSTM scan
  zero_u32_n<<<1, 64, 0, stream>>>(flags, 64);
  const size_t smem =
      (size_t)(64 * WROWU) * 4 + (size_t)(1024 + 256 + 64 + 64 + 16) * 4;
  lstm_scan<<<64, 256, smem, stream>>>((const unsigned int*)Whhb, GX, bhh, HS,
                                       flags, S_STEPS);

  // Head
  head_kernel<<<S_STEPS, 256, 0, stream>>>(HS, whead, bhead, out);
}
